// GIN_19421842112604
// MI455X (gfx1250) — matrix operand order repaired and checked by reading, not compiled, not run on hardware
//
#include <hip/hip_runtime.h>
#include <hip/hip_bf16.h>
#include <math.h>

// ---------------- problem constants (from reference) ----------------
#define NN   20000
#define EE   320000
#define DIN  128
#define HH   1024
#define DMID 256
#define DOUT 10
#define BN_EPS 1e-5f

typedef __attribute__((ext_vector_type(16))) __bf16 v16bf;
typedef __attribute__((ext_vector_type(8)))  __bf16 v8bf;
typedef __attribute__((ext_vector_type(4)))  __bf16 v4bf;
typedef __attribute__((ext_vector_type(8)))  float  v8f;
typedef __attribute__((ext_vector_type(4)))  float  v4f;

// low 32 bits of a generic pointer into LDS = LDS byte offset (ISA 10.2 aperture map)
__device__ __forceinline__ unsigned lds_off(const void* p) {
  return (unsigned)(unsigned long long)p;
}

// one 16-byte async DMA chunk: global -> LDS, tracked by ASYNCcnt (no VGPR data path)
__device__ __forceinline__ void async_copy_b128(unsigned lds, const void* g) {
  asm volatile("global_load_async_to_lds_b128 %0, %1, off" : : "v"(lds), "v"(g) : "memory");
}
__device__ __forceinline__ void wait_asynccnt0() {
  asm volatile("s_wait_asynccnt 0x0" ::: "memory");
}

// ---------------- elementwise copy (agg init: agg = h) ----------------
__global__ __launch_bounds__(256) void k_copy_f4(const float* __restrict__ src,
                                                 float* __restrict__ dst, int n4) {
  int i = blockIdx.x * blockDim.x + threadIdx.x;
  if (i < n4) ((v4f*)dst)[i] = ((const v4f*)src)[i];
}

// ---------------- fp32 -> bf16 activation conversion ----------------
__global__ __launch_bounds__(256) void k_cvt_bf16(const float* __restrict__ in,
                                                  __bf16* __restrict__ out, int n4) {
  int i = blockIdx.x * blockDim.x + threadIdx.x;
  if (i >= n4) return;
  v4f v = ((const v4f*)in)[i];
  v4bf b;
  b[0] = (__bf16)v[0]; b[1] = (__bf16)v[1];
  b[2] = (__bf16)v[2]; b[3] = (__bf16)v[3];
  ((v4bf*)out)[i] = b;
}

// ---------------- weight transpose + convert: WT[n][k] = bf16(W[k][n]) ----------------
__global__ __launch_bounds__(256) void k_wt_cvt(const float* __restrict__ W,
                                                __bf16* __restrict__ WT, int K, int Nout) {
  int idx = blockIdx.x * 256 + threadIdx.x;
  if (idx >= K * Nout) return;
  int n = idx / K, k = idx - n * K;
  WT[idx] = (__bf16)W[(long long)k * Nout + n];
}

// ---------------- edge scatter-add: agg[dst] += x[src] (L2-resident fp32) ----------------
__global__ __launch_bounds__(256) void k_scatter(const float* __restrict__ x,
                                                 const int* __restrict__ srcI,
                                                 const int* __restrict__ dstI,
                                                 float* agg, int E, int D4) {
  long long i = (long long)blockIdx.x * blockDim.x + threadIdx.x;
  long long tot = (long long)E * D4;
  if (i >= tot) return;
  int e = (int)(i / D4);
  int c = (int)(i - (long long)e * D4) * 4;
  int s = srcI[e], d = dstI[e];
  v4f v = *(const v4f*)(x + (long long)s * D4 * 4 + c);
  float* p = agg + (long long)d * D4 * 4 + c;
  unsafeAtomicAdd(p + 0, v[0]);   // -> global_atomic_add_f32
  unsafeAtomicAdd(p + 1, v[1]);
  unsafeAtomicAdd(p + 2, v[2]);
  unsafeAtomicAdd(p + 3, v[3]);
}

// ---------------- async-staged WMMA bf16 GEMM ----------------
// C = [relu](A @ W + bias);  A: MxK bf16 row-major, WT: NoutxK bf16 row-major
// (pre-transposed), C: fp32 or bf16.  Block tile 128x128, BK=32, double-buffered
// LDS staged via GLOBAL_LOAD_ASYNC_TO_LDS_B128.  8 waves = 4(M) x 2(N); each wave
// runs 2x4 v_wmma_f32_16x16x32_bf16 tiles with f32 accumulation.

// stage one 128x32-bf16 tile (row-major, row stride ldK elems) into LDS
// (96-byte LDS row stride).  512 16B chunks over 256 threads = 2 async issues/thread.
__device__ __forceinline__ void stage_tile(const __bf16* __restrict__ gbase, int ldK,
                                           int rowsAvail, unsigned ldsBase, int tid,
                                           int k0) {
#pragma unroll
  for (int p = 0; p < 2; p++) {
    int q = p * 256 + tid;          // 0..511
    int row = q >> 2;               // 0..127
    int ch = (q & 3) << 4;          // byte chunk: 0,16,32,48
    int r = row < rowsAvail ? row : (rowsAvail - 1);   // clamp (no divergence)
    const char* g = (const char*)(gbase + (long long)r * ldK + k0) + ch;
    async_copy_b128(ldsBase + row * 96u + (unsigned)ch, g);
  }
}

template <int RELU, int OUTBF16>
__global__ __launch_bounds__(256) void k_gemm(const __bf16* __restrict__ A,
                                              const __bf16* __restrict__ WT,
                                              const float* __restrict__ bias,
                                              void* __restrict__ Cout,
                                              int M, int K, int Nout) {
  // 48-bf16 (96 B) row stride keeps ds_load_b128 16B-aligned; x2 for double buffer.
  __shared__ __align__(16) __bf16 As[2][128][48];
  __shared__ __align__(16) __bf16 Bt[2][128][48];

  const int tid   = threadIdx.x;
  const int lane  = tid & 31;
  const int wave  = tid >> 5;
  const int waveM = wave & 3;
  const int waveN = wave >> 2;
  const int m0 = blockIdx.x * 128;
  const int n0 = blockIdx.y * 128;

  const __bf16* Ab = A + (long long)m0 * K;
  const __bf16* Wb = WT + (long long)n0 * K;
  const int mAvail = M - m0;
  const int nAvail = Nout - n0;

  const unsigned asB[2] = { lds_off(&As[0][0][0]), lds_off(&As[1][0][0]) };
  const unsigned btB[2] = { lds_off(&Bt[0][0][0]), lds_off(&Bt[1][0][0]) };

  v8f acc[2][4];
#pragma unroll
  for (int i = 0; i < 2; i++)
#pragma unroll
    for (int j = 0; j < 4; j++) acc[i][j] = {};

  // prologue: stage first K-tile into buffer 0
  stage_tile(Ab, K, mAvail, asB[0], tid, 0);
  stage_tile(Wb, K, nAvail, btB[0], tid, 0);

  const int nk = K >> 5;
  for (int ik = 0; ik < nk; ik++) {
    const int b = ik & 1;
    wait_asynccnt0();       // own async DMAs into buffer b done
    __syncthreads();        // all waves' DMAs done; prev reads of b^1 done

    if (ik + 1 < nk) {      // overlap: stage next tile while computing this one
      stage_tile(Ab, K, mAvail, asB[b ^ 1], tid, (ik + 1) * 32);
      stage_tile(Wb, K, nAvail, btB[b ^ 1], tid, (ik + 1) * 32);
    }

    // ---- fragments per ISA VGPR layouts ----
    // A 16x32: lanes 0-15 hold K {0..7,16..23}; lanes 16-31 {8..15,24..31}
    const int kqoff = (lane >> 4) * 8;
    // B 32x16: lanes 0-15 hold K {0..7,16..23} of column N=lane&15; lanes 16-31 {8..15,24..31}
    const int kb = (lane >> 4) * 8;

    v16bf afrag[2], bfrag[4];
#pragma unroll
    for (int mt = 0; mt < 2; mt++) {
      const __bf16* rp = &As[b][waveM * 32 + mt * 16 + (lane & 15)][0];
      v8bf lo = *(const v8bf*)(rp + kqoff);
      v8bf hi = *(const v8bf*)(rp + 16 + kqoff);
      union { v16bf v; v8bf h[2]; } u; u.h[0] = lo; u.h[1] = hi;
      afrag[mt] = u.v;
    }
#pragma unroll
    for (int nt = 0; nt < 4; nt++) {
      const __bf16* cp = &Bt[b][waveN * 64 + nt * 16 + (lane & 15)][0];
      v8bf lo = *(const v8bf*)(cp + kb);
      v8bf hi = *(const v8bf*)(cp + kb + 16);
      union { v16bf v; v8bf h[2]; } u; u.h[0] = lo; u.h[1] = hi;
      bfrag[nt] = u.v;
    }

#pragma unroll
    for (int mt = 0; mt < 2; mt++)
#pragma unroll
      for (int nt = 0; nt < 4; nt++)
        acc[mt][nt] = __builtin_amdgcn_wmma_f32_16x16x32_bf16(
            false, afrag[mt], false, bfrag[nt], (short)0, acc[mt][nt],
            false, false);
  }

  // ---- epilogue: bias (+ReLU); C/D layout: VGPR r -> M=r+8*(lane>=16), N=lane&15 ----
#pragma unroll
  for (int mt = 0; mt < 2; mt++) {
#pragma unroll
    for (int nt = 0; nt < 4; nt++) {
      int gcol = n0 + waveN * 64 + nt * 16 + (lane & 15);
      if (gcol >= Nout) continue;
      float bv = bias[gcol];
      int mbase = m0 + waveM * 32 + mt * 16 + ((lane >> 4) << 3);
#pragma unroll
      for (int r = 0; r < 8; r++) {
        int gm = mbase + r;
        if (gm < M) {
          float v = acc[mt][nt][r] + bv;
          if (RELU) v = fmaxf(v, 0.f);
          long long idx = (long long)gm * Nout + gcol;
          if (OUTBF16) ((__bf16*)Cout)[idx] = (__bf16)v;
          else         ((float*)Cout)[idx] = v;
        }
      }
    }
  }
}

// ---------------- BatchNorm (training-mode, biased var) ----------------
__global__ __launch_bounds__(256) void k_zero(float* p, int n) {
  int i = blockIdx.x * blockDim.x + threadIdx.x;
  if (i < n) p[i] = 0.f;
}

__global__ __launch_bounds__(256) void k_bn_partial(const float* __restrict__ h,
                                                    float* sum, float* sumsq,
                                                    int Nrows, int rowsPer) {
  int col = blockIdx.x * 256 + threadIdx.x;     // grid.x = HH/256
  int r0 = blockIdx.y * rowsPer;
  int r1 = r0 + rowsPer; if (r1 > Nrows) r1 = Nrows;
  float s = 0.f, ss = 0.f;
  for (int r = r0; r < r1; r++) {
    float v = h[(long long)r * HH + col];
    s += v; ss += v * v;
  }
  unsafeAtomicAdd(&sum[col], s);
  unsafeAtomicAdd(&sumsq[col], ss);
}

// fold gamma/beta into scale/shift: y = scale*x + shift
__global__ __launch_bounds__(256) void k_bn_finalize(const float* sum, const float* sumsq,
                                                     float* scale, float* shift,
                                                     const float* gamma, const float* beta,
                                                     float Ninv) {
  int c = blockIdx.x * 256 + threadIdx.x;
  float m   = sum[c] * Ninv;
  float var = sumsq[c] * Ninv - m * m;
  float rstd = rsqrtf(var + BN_EPS);
  float sc = gamma[c] * rstd;
  scale[c] = sc;
  shift[c] = beta[c] - sc * m;
}

__global__ __launch_bounds__(256) void k_bn_apply_relu(const float* __restrict__ in,
                                                       float* __restrict__ out,
                                                       const float* __restrict__ scale,
                                                       const float* __restrict__ shift,
                                                       int n4) {
  int i = blockIdx.x * blockDim.x + threadIdx.x;
  if (i >= n4) return;
  v4f v = ((const v4f*)in)[i];
  int c = (i * 4) & (HH - 1);
#pragma unroll
  for (int j = 0; j < 4; j++) v[j] = fmaxf(v[j] * scale[c + j] + shift[c + j], 0.f);
  ((v4f*)out)[i] = v;
}

// ---------------- head: out = log_softmax(mid @ Wl2 + bl2) ----------------
__global__ __launch_bounds__(256) void k_head(const float* __restrict__ mid,
                                              const float* __restrict__ Wl2,
                                              const float* __restrict__ bl2,
                                              float* __restrict__ out, int Nn) {
  __shared__ float sW[DMID * DOUT];
  __shared__ float sB[DOUT];
  int tid = threadIdx.x;
  for (int i = tid; i < DMID * DOUT; i += 256) sW[i] = Wl2[i];
  if (tid < DOUT) sB[tid] = bl2[tid];
  __syncthreads();

  int wave = tid >> 5, lane = tid & 31;
  int node = blockIdx.x * 8 + wave;
  if (node >= Nn) return;

  float p[DOUT];
#pragma unroll
  for (int o = 0; o < DOUT; o++) p[o] = 0.f;
#pragma unroll
  for (int i = 0; i < DMID / 32; i++) {
    int k = lane + 32 * i;
    float m = mid[(long long)node * DMID + k];
    const float* wr = &sW[k * DOUT];
#pragma unroll
    for (int o = 0; o < DOUT; o++) p[o] += m * wr[o];
  }
#pragma unroll
  for (int o = 0; o < DOUT; o++)
    for (int off = 16; off; off >>= 1) p[o] += __shfl_xor(p[o], off, 32);

  if (lane == 0) {
    float v[DOUT], mx = -INFINITY;
#pragma unroll
    for (int o = 0; o < DOUT; o++) { v[o] = p[o] + sB[o]; mx = fmaxf(mx, v[o]); }
    float s = 0.f;
#pragma unroll
    for (int o = 0; o < DOUT; o++) s += __expf(v[o] - mx);
    float lse = mx + __logf(s);
#pragma unroll
    for (int o = 0; o < DOUT; o++) out[(long long)node * DOUT + o] = v[o] - lse;
  }
}

// ---------------- host-side launch ----------------
static inline int cdiv(long long a, long long b) { return (int)((a + b - 1) / b); }

extern "C" void kernel_launch(void* const* d_in, const int* in_sizes, int n_in,
                              void* d_out, int out_size, void* d_ws, size_t ws_size,
                              hipStream_t stream) {
  const float* x   = (const float*)d_in[0];
  const int*   ei  = (const int*)d_in[1];
  const int*   src = ei;
  const int*   dst = ei + EE;
  const float* W1a = (const float*)d_in[2];  const float* b1a = (const float*)d_in[3];
  const float* W1b = (const float*)d_in[4];  const float* b1b = (const float*)d_in[5];
  const float* g1  = (const float*)d_in[6];  const float* be1 = (const float*)d_in[7];
  const float* W2a = (const float*)d_in[8];  const float* b2a = (const float*)d_in[9];
  const float* W2b = (const float*)d_in[10]; const float* b2b = (const float*)d_in[11];
  const float* g2  = (const float*)d_in[12]; const float* be2 = (const float*)d_in[13];
  const float* Wl1 = (const float*)d_in[14]; const float* bl1 = (const float*)d_in[15];
  const float* Wl2 = (const float*)d_in[16]; const float* bl2 = (const float*)d_in[17];
  float* outp = (float*)d_out;

  // ---- workspace layout ----
  float*  bufA = (float*)d_ws;                          // N x 1024 fp32
  float*  bufB = bufA + (size_t)NN * HH;                // N x 1024 fp32
  __bf16* hb   = (__bf16*)(bufB + (size_t)NN * HH);     // N x 1024 bf16 (GEMM A)
  __bf16* tb   = hb + (size_t)NN * HH;                  // N x 1024 bf16 (GEMM out)
  __bf16* w1aT = tb + (size_t)NN * HH;                  // HH x DIN
  __bf16* w1bT = w1aT + (size_t)HH * DIN;               // HH x HH
  __bf16* w2aT = w1bT + (size_t)HH * HH;                // HH x HH
  __bf16* w2bT = w2aT + (size_t)HH * HH;                // HH x HH
  __bf16* wl1T = w2bT + (size_t)HH * HH;                // DMID x HH
  float*  s0   = (float*)(wl1T + (size_t)DMID * HH);    // sums   (HH)
  float*  s1   = s0 + HH;                               // sumsq  (HH)
  float*  s2   = s1 + HH;                               // scale  (HH)
  float*  s3   = s2 + HH;                               // shift  (HH)

  const dim3 B256(256);
  const int gM = cdiv(NN, 128);                         // 157

  // ---- one-time weight transpose+convert (tiny; L2-resident afterwards) ----
  k_wt_cvt<<<cdiv((long long)DIN * HH, 256), B256, 0, stream>>>(W1a, w1aT, DIN, HH);
  k_wt_cvt<<<cdiv((long long)HH * HH, 256), B256, 0, stream>>>(W1b, w1bT, HH, HH);
  k_wt_cvt<<<cdiv((long long)HH * HH, 256), B256, 0, stream>>>(W2a, w2aT, HH, HH);
  k_wt_cvt<<<cdiv((long long)HH * HH, 256), B256, 0, stream>>>(W2b, w2bT, HH, HH);
  k_wt_cvt<<<cdiv((long long)HH * DMID, 256), B256, 0, stream>>>(Wl1, wl1T, HH, DMID);

  // ---- layer 1: h0 = x + segment_sum(x[src], dst) ----
  k_copy_f4<<<cdiv((long long)NN * DIN / 4, 256), B256, 0, stream>>>(x, bufA, NN * DIN / 4);
  k_scatter<<<cdiv((long long)EE * (DIN / 4), 256), B256, 0, stream>>>(x, src, dst, bufA, EE, DIN / 4);
  k_cvt_bf16<<<cdiv((long long)NN * DIN / 4, 256), B256, 0, stream>>>(bufA, hb, NN * DIN / 4);
  // t1 = relu(h0 @ W1a + b1a)  [bf16 out]
  k_gemm<1, 1><<<dim3(gM, HH / 128), B256, 0, stream>>>(hb, w1aT, b1a, tb, NN, DIN, HH);
  // h1 = t1 @ W1b + b1b        [fp32 out]
  k_gemm<0, 0><<<dim3(gM, HH / 128), B256, 0, stream>>>(tb, w1bT, b1b, bufA, NN, HH, HH);
  // BN1 + relu -> bufB
  k_zero<<<cdiv(2 * HH, 256), B256, 0, stream>>>(s0, 2 * HH);
  k_bn_partial<<<dim3(HH / 256, 80), B256, 0, stream>>>(bufA, s0, s1, NN, 250);
  k_bn_finalize<<<HH / 256, B256, 0, stream>>>(s0, s1, s2, s3, g1, be1, 1.0f / NN);
  k_bn_apply_relu<<<cdiv((long long)NN * HH / 4, 256), B256, 0, stream>>>(bufA, bufB, s2, s3, NN * HH / 4);

  // ---- layer 2: h2in = h + segment_sum(h[src], dst) ----
  k_copy_f4<<<cdiv((long long)NN * HH / 4, 256), B256, 0, stream>>>(bufB, bufA, NN * HH / 4);
  k_scatter<<<cdiv((long long)EE * (HH / 4), 256), B256, 0, stream>>>(bufB, src, dst, bufA, EE, HH / 4);
  k_cvt_bf16<<<cdiv((long long)NN * HH / 4, 256), B256, 0, stream>>>(bufA, hb, NN * HH / 4);
  k_gemm<1, 1><<<dim3(gM, HH / 128), B256, 0, stream>>>(hb, w2aT, b2a, tb, NN, HH, HH);
  k_gemm<0, 0><<<dim3(gM, HH / 128), B256, 0, stream>>>(tb, w2bT, b2b, bufA, NN, HH, HH);
  k_zero<<<cdiv(2 * HH, 256), B256, 0, stream>>>(s0, 2 * HH);
  k_bn_partial<<<dim3(HH / 256, 80), B256, 0, stream>>>(bufA, s0, s1, NN, 250);
  k_bn_finalize<<<HH / 256, B256, 0, stream>>>(s0, s1, s2, s3, g2, be2, 1.0f / NN);
  k_bn_apply_relu<<<cdiv((long long)NN * HH / 4, 256), B256, 0, stream>>>(bufA, bufB, s2, s3, NN * HH / 4);

  // ---- MLP head ----
  k_cvt_bf16<<<cdiv((long long)NN * HH / 4, 256), B256, 0, stream>>>(bufB, hb, NN * HH / 4);
  k_gemm<1, 0><<<dim3(gM, DMID / 128), B256, 0, stream>>>(hb, wl1T, bl1, bufA, NN, HH, DMID);
  k_head<<<cdiv(NN, 8), B256, 0, stream>>>(bufA, Wl2, bl2, outp, NN);
}
